// LinearMultiHeadAttention_13185549599045
// MI455X (gfx1250) — matrix operand order repaired and checked by reading, not compiled, not run on hardware
//
#include <hip/hip_runtime.h>

typedef __attribute__((ext_vector_type(16))) __bf16 v16bf;
typedef __attribute__((ext_vector_type(8)))  float  v8f;
typedef __attribute__((ext_vector_type(4)))  float  f32x4;

#define EPS 1e-6f

// Reinterpret two contiguous 16B chunks as one 16x(bf16) WMMA fragment.
union FragU { f32x4 f4[2]; v16bf v; };

static __device__ __forceinline__ v16bf load_frag16(const __bf16* p0,
                                                    const __bf16* p1) {
  FragU u;
  u.f4[0] = *(const f32x4*)p0;
  u.f4[1] = *(const f32x4*)p1;
  return u.v;
}

// ---------------------------------------------------------------------------
// f32 -> bf16 bulk convert (weights)
// ---------------------------------------------------------------------------
__global__ void cvt_f32_bf16(const float* __restrict__ src,
                             __bf16* __restrict__ dst, int n) {
  int i = (blockIdx.x * blockDim.x + threadIdx.x) * 8;
  if (i >= n) return;
  f32x4 a = *(const f32x4*)(src + i);
  f32x4 b = *(const f32x4*)(src + i + 4);
  __bf16 t[8];
  t[0] = (__bf16)a.x; t[1] = (__bf16)a.y; t[2] = (__bf16)a.z; t[3] = (__bf16)a.w;
  t[4] = (__bf16)b.x; t[5] = (__bf16)b.y; t[6] = (__bf16)b.z; t[7] = (__bf16)b.w;
  *(f32x4*)&dst[i] = *(f32x4*)&t[0];
}

// ---------------------------------------------------------------------------
// Tiled GEMM: C[M,N] = act(A[M,K] @ W[N,K]^T + bias[N])
//   OUT_MODE: 0 = bf16 natural [m][n], 1 = f32 natural, 2 = bf16 transposed
//   [n][m] (packed b128 stores: lane's 8 acc elements are consecutive m).
//   Block = 128 threads = 4 waves; block tile 128x64; wave tile 32x64.
// ---------------------------------------------------------------------------
template <bool A_BF16, bool DO_ELU, int OUT_MODE>
__global__ void __launch_bounds__(128)
gemm_bias_act(const float* __restrict__ Af32, const __bf16* __restrict__ Abf,
              const __bf16* __restrict__ Wb, const float* __restrict__ bias,
              __bf16* __restrict__ Cb, float* __restrict__ Cf,
              int M, int N, int K) {
  __shared__ __bf16 As[128][40];  // [m][k], 80B rows: conflict-free b128
  __shared__ __bf16 Bs[64][40];   // [n][k]

  const int tid  = threadIdx.x;
  const int lane = tid & 31;
  const int wave = tid >> 5;  // 0..3
  const int n0   = blockIdx.x * 64;
  const int m0   = blockIdx.y * 128;

  v8f zero8 = {};
  v8f acc[2][4];
#pragma unroll
  for (int i = 0; i < 2; ++i)
#pragma unroll
    for (int j = 0; j < 4; ++j) acc[i][j] = zero8;

  for (int kk = 0; kk < K; kk += 32) {
    // ---- stage A tile (128 x 32): one full row per thread ----
    if (A_BF16) {
      const __bf16* src = Abf + (size_t)(m0 + tid) * K + kk;
#pragma unroll
      for (int i = 0; i < 32; i += 8)
        *(f32x4*)&As[tid][i] = *(const f32x4*)(src + i);
      if (kk + 32 < K) __builtin_prefetch(src + 32, 0, 1);
    } else {
      const float* src = Af32 + (size_t)(m0 + tid) * K + kk;
      __bf16 tmp[32];
#pragma unroll
      for (int i = 0; i < 32; i += 4) {
        f32x4 f = *(const f32x4*)(src + i);
        tmp[i + 0] = (__bf16)f.x;
        tmp[i + 1] = (__bf16)f.y;
        tmp[i + 2] = (__bf16)f.z;
        tmp[i + 3] = (__bf16)f.w;
      }
#pragma unroll
      for (int i = 0; i < 32; i += 8)
        *(f32x4*)&As[tid][i] = *(f32x4*)&tmp[i];
      if (kk + 32 < K) __builtin_prefetch(src + 32, 0, 1);
    }
    // ---- stage B tile (64 x 32): Bs[n][k] = Wb[n0+n][kk+k] ----
    {
      const int rB = tid >> 1;
      const int cB = (tid & 1) * 16;
      const __bf16* wsrc = Wb + (size_t)(n0 + rB) * K + kk + cB;
      *(f32x4*)&Bs[rB][cB]     = *(const f32x4*)(wsrc);
      *(f32x4*)&Bs[rB][cB + 8] = *(const f32x4*)(wsrc + 8);
      if (kk + 32 < K) __builtin_prefetch(wsrc + 32, 0, 1);
    }
    __syncthreads();

    const int kb  = (lane < 16) ? 0 : 8;
    const int kb2 = (lane < 16) ? 0 : 16;
    const int c15 = lane & 15;
    v16bf a[2];
#pragma unroll
    for (int mi = 0; mi < 2; ++mi) {
      const int row = wave * 32 + mi * 16 + c15;
      a[mi] = load_frag16(&As[row][kb], &As[row][kb + 16]);
    }
#pragma unroll
    for (int nt = 0; nt < 4; ++nt) {
      const int n = nt * 16 + c15;
      v16bf bfrag = load_frag16(&Bs[n][kb], &Bs[n][kb + 16]);
#pragma unroll
      for (int mi = 0; mi < 2; ++mi)
        acc[mi][nt] = __builtin_amdgcn_wmma_f32_16x16x32_bf16(
            false, a[mi], false, bfrag, (short)0, acc[mi][nt], false, false);
    }
    __syncthreads();
  }

  // ---- epilogue ----
  const int c15 = lane & 15;
#pragma unroll
  for (int mi = 0; mi < 2; ++mi) {
#pragma unroll
    for (int nt = 0; nt < 4; ++nt) {
      const int n = n0 + nt * 16 + c15;
      const float bv = bias[n];
      if (OUT_MODE == 2) {
        // transposed bf16 [n][m]: 8 consecutive m -> one packed b128 store
        __bf16 tmp[8];
#pragma unroll
        for (int rr = 0; rr < 8; ++rr) {
          float v = acc[mi][nt][rr] + bv;
          if (DO_ELU) v = (v > 0.0f) ? (v + 1.0f) : __expf(v);
          tmp[rr] = (__bf16)v;
        }
        const int mstart = m0 + wave * 32 + mi * 16 + 8 * (lane >> 4);
        *(f32x4*)&Cb[(size_t)n * M + mstart] = *(f32x4*)&tmp[0];
      } else {
#pragma unroll
        for (int rr = 0; rr < 8; ++rr) {
          int m = m0 + wave * 32 + mi * 16 + rr + 8 * (lane >> 4);
          float v = acc[mi][nt][rr] + bv;
          if (DO_ELU) v = (v > 0.0f) ? (v + 1.0f) : __expf(v);
          if (OUT_MODE == 1) Cf[(size_t)m * N + n] = v;
          else               Cb[(size_t)m * N + n] = (__bf16)v;
        }
      }
    }
  }
}

// ---------------------------------------------------------------------------
// KV reduction, fully LDS-free: Kt[d][s] and Vt[e][s] are head-transposed, so
// both WMMA operands are contiguous 16B global runs. ksum is fused as a
// constant ones-column fragment (n==64). acc = 4x5 tiles; atomic f32 reduce.
//   kvacc layout: [bh][d][80].
// ---------------------------------------------------------------------------
#define SCHUNK 1024
__global__ void __launch_bounds__(32)
kv_reduce(const __bf16* __restrict__ Kt, const __bf16* __restrict__ Vt,
          float* __restrict__ kvacc, int S, int Mtot) {
  const int lane = threadIdx.x;
  const int b = blockIdx.z, h = blockIdx.y;
  const int s0base = blockIdx.x * SCHUNK;
  const __bf16* KtB = Kt + (size_t)(h * 64) * Mtot + (size_t)b * S;
  const __bf16* VtB = Vt + (size_t)(h * 64) * Mtot + (size_t)b * S;

  const int kb  = (lane < 16) ? 0 : 8;
  const int kb2 = (lane < 16) ? 0 : 16;
  const int c15 = lane & 15;

  // constant ones-column fragment: column n==64 (lane c15==0) is all 1.0
  v16bf ones = {};
  if (c15 == 0) {
#pragma unroll
    for (int i = 0; i < 16; ++i) ones[i] = (__bf16)1.0f;
  }

  v8f zero8 = {};
  v8f acc[4][5];
#pragma unroll
  for (int i = 0; i < 4; ++i)
#pragma unroll
    for (int j = 0; j < 5; ++j) acc[i][j] = zero8;

  for (int s0 = s0base; s0 < s0base + SCHUNK; s0 += 32) {
    v16bf a[4], bv[4];
#pragma unroll
    for (int dt = 0; dt < 4; ++dt) {
      const __bf16* p = KtB + (size_t)(dt * 16 + c15) * Mtot + s0;
      a[dt] = load_frag16(p + kb, p + kb + 16);
    }
#pragma unroll
    for (int et = 0; et < 4; ++et) {
      const __bf16* p = VtB + (size_t)(et * 16 + c15) * Mtot + s0;
      bv[et] = load_frag16(p + kb, p + kb + 16);
    }
#pragma unroll
    for (int dt = 0; dt < 4; ++dt) {
#pragma unroll
      for (int et = 0; et < 4; ++et)
        acc[dt][et] = __builtin_amdgcn_wmma_f32_16x16x32_bf16(
            false, a[dt], false, bv[et], (short)0, acc[dt][et], false, false);
      acc[dt][4] = __builtin_amdgcn_wmma_f32_16x16x32_bf16(
          false, a[dt], false, ones, (short)0, acc[dt][4], false, false);
    }
  }

  float* kvb = kvacc + (size_t)(b * 16 + h) * 64 * 80;
#pragma unroll
  for (int dt = 0; dt < 4; ++dt)
#pragma unroll
    for (int et = 0; et < 5; ++et)
#pragma unroll
      for (int rr = 0; rr < 8; ++rr) {
        int d = dt * 16 + rr + 8 * (lane >> 4);
        int e = et * 16 + c15;
        atomicAdd(&kvb[d * 80 + e], acc[dt][et][rr]);
      }
}

// ---------------------------------------------------------------------------
// kvacc[bh][d][80] (f32) -> kvxT[bh][e][d] (bf16). Row e==64 carries ksum.
// ---------------------------------------------------------------------------
__global__ void build_kvxT(const float* __restrict__ kvacc,
                           __bf16* __restrict__ kvxT) {
  int bh = blockIdx.x;
  const float* src = kvacc + (size_t)bh * 64 * 80;
  __bf16* o = kvxT + (size_t)bh * 80 * 64;
  for (int idx = threadIdx.x; idx < 80 * 64; idx += blockDim.x) {
    int e = idx >> 6, d = idx & 63;
    o[idx] = (__bf16)src[d * 80 + e];
  }
}

// ---------------------------------------------------------------------------
// attn, LDS-free via role swap: C[e][t] = KVxT[80x64] @ Qf^T.
//   A = kvxT rows (natural, L1-resident), B = Qf rows (natural). Output per
//   lane is e-major -> packed b128 stores. Normalizer = row e==64 (tile 4),
//   broadcast across halves with one shfl. Block = 4 waves x 16 t = 64 t.
// ---------------------------------------------------------------------------
__global__ void __launch_bounds__(128)
attn_kernel(const __bf16* __restrict__ Qf, const __bf16* __restrict__ kvxT,
            __bf16* __restrict__ attn, int S) {
  const int lane = threadIdx.x & 31;
  const int wave = threadIdx.x >> 5;
  const int b = blockIdx.z, h = blockIdx.y;
  const int t0 = blockIdx.x * 64;
  const int c15 = lane & 15;
  const int trow = t0 + wave * 16 + c15;

  const __bf16* KVT  = kvxT + (size_t)(b * 16 + h) * 80 * 64;
  const __bf16* qrow = Qf + ((size_t)b * S + trow) * 1024 + h * 64;

  v8f zero8 = {};
  v8f acc[5];
#pragma unroll
  for (int i = 0; i < 5; ++i) acc[i] = zero8;

#pragma unroll
  for (int kt = 0; kt < 2; ++kt) {
    const int kb2 = kt * 32 + ((lane < 16) ? 0 : 8);
    v16bf bq = load_frag16(qrow + kb2, qrow + kb2 + 16);
    const int kb = kt * 32 + ((lane < 16) ? 0 : 8);
#pragma unroll
    for (int mt = 0; mt < 5; ++mt) {
      const __bf16* ar = KVT + (size_t)(mt * 16 + c15) * 64 + kb;
      v16bf a = load_frag16(ar, ar + 16);
      acc[mt] = __builtin_amdgcn_wmma_f32_16x16x32_bf16(
          false, a, false, bq, (short)0, acc[mt], false, false);
    }
  }

  // normalizer: (m=64,n=t) lives in lanes 0..15 of tile 4, rr=0
  float dot = __shfl(acc[4][0], c15, 32);
  float nrm = 1.0f / (dot + EPS);

  __bf16* orow = attn + ((size_t)b * S + trow) * 1024 + h * 64 + 8 * (lane >> 4);
#pragma unroll
  for (int mt = 0; mt < 4; ++mt) {
    __bf16 tmp[8];
#pragma unroll
    for (int rr = 0; rr < 8; ++rr) tmp[rr] = (__bf16)(acc[mt][rr] * nrm);
    *(f32x4*)&orow[mt * 16] = *(f32x4*)&tmp[0];
  }
}

__global__ void zero_f32(float* __restrict__ p, int n) {
  int i = blockIdx.x * blockDim.x + threadIdx.x;
  if (i < n) p[i] = 0.0f;
}

// ---------------------------------------------------------------------------
extern "C" void kernel_launch(void* const* d_in, const int* in_sizes, int n_in,
                              void* d_out, int out_size, void* d_ws, size_t ws_size,
                              hipStream_t stream) {
  const float* query = (const float*)d_in[0];
  const float* key_  = (const float*)d_in[1];
  const float* value = (const float*)d_in[2];
  const float* Wq = (const float*)d_in[3];
  const float* bq = (const float*)d_in[4];
  const float* Wk = (const float*)d_in[5];
  const float* bk = (const float*)d_in[6];
  const float* Wv = (const float*)d_in[7];
  const float* bv = (const float*)d_in[8];
  const float* Wo = (const float*)d_in[9];
  const float* bo = (const float*)d_in[10];
  float* out = (float*)d_out;

  const int B = 4, S = 8192, D = 1024;
  const int M = B * S;  // 32768

  // workspace carve-up (~268 MB)
  char* ws = (char*)d_ws;
  __bf16* Qf   = (__bf16*)ws; ws += (size_t)M * D * 2;  // [t][d] natural
  __bf16* Kt   = (__bf16*)ws; ws += (size_t)M * D * 2;  // [d_global][m] transposed
  __bf16* Vt   = (__bf16*)ws; ws += (size_t)M * D * 2;  // [e_global][m] transposed
  __bf16* attn = (__bf16*)ws; ws += (size_t)M * D * 2;  // [t][h*64+e] natural
  __bf16* Wqb  = (__bf16*)ws; ws += (size_t)D * D * 2;
  __bf16* Wkb  = (__bf16*)ws; ws += (size_t)D * D * 2;
  __bf16* Wvb  = (__bf16*)ws; ws += (size_t)D * D * 2;
  __bf16* Wob  = (__bf16*)ws; ws += (size_t)D * D * 2;
  float*  kvacc = (float*)ws; ws += (size_t)B * 16 * 64 * 80 * 4;
  __bf16* kvxT = (__bf16*)ws;

  // 0) weights -> bf16 (L2-resident, reused by every tile)
  const int wn = D * D;
  cvt_f32_bf16<<<wn / 8 / 256, 256, 0, stream>>>(Wq, Wqb, wn);
  cvt_f32_bf16<<<wn / 8 / 256, 256, 0, stream>>>(Wk, Wkb, wn);
  cvt_f32_bf16<<<wn / 8 / 256, 256, 0, stream>>>(Wv, Wvb, wn);
  cvt_f32_bf16<<<wn / 8 / 256, 256, 0, stream>>>(Wo, Wob, wn);

  dim3 gGemm(D / 64, M / 128);  // (16, 256)

  // 1) projections: Q natural; K,V head-transposed (free transpose in epilogue)
  gemm_bias_act<false, true,  0><<<gGemm, 128, 0, stream>>>(
      query, nullptr, Wqb, bq, Qf, nullptr, M, D, D);
  gemm_bias_act<false, true,  2><<<gGemm, 128, 0, stream>>>(
      key_,  nullptr, Wkb, bk, Kt, nullptr, M, D, D);
  gemm_bias_act<false, false, 2><<<gGemm, 128, 0, stream>>>(
      value, nullptr, Wvb, bv, Vt, nullptr, M, D, D);

  // 2) KV + ksum in one LDS-free WMMA reduction
  const int nz = B * 16 * 64 * 80;
  zero_f32<<<(nz + 255) / 256, 256, 0, stream>>>(kvacc, nz);
  kv_reduce<<<dim3(S / SCHUNK, 16, B), 32, 0, stream>>>(Kt, Vt, kvacc, S, M);

  // 3) transposed bf16 augmented KV (row 64 = ksum)
  build_kvxT<<<B * 16, 128, 0, stream>>>(kvacc, kvxT);

  // 4) attn = (KVx @ Q^T)^T with fused 1/(q.ksum+eps), packed b128 stores
  attn_kernel<<<dim3(S / 64, 16, B), 128, 0, stream>>>(Qf, kvxT, attn, S);

  // 5) output projection -> f32
  gemm_bias_act<true, false, 1><<<gGemm, 128, 0, stream>>>(
      nullptr, attn, Wob, bo, nullptr, out, M, D, D);
}
